// GaussianLearner_46462956208560
// MI455X (gfx1250) — matrix operand order repaired and checked by reading, not compiled, not run on hardware
//
#include <hip/hip_runtime.h>
#include <cmath>

typedef __attribute__((ext_vector_type(16))) _Float16 v16h;
typedef __attribute__((ext_vector_type(8)))  float    v8f;
typedef __attribute__((ext_vector_type(4)))  unsigned v4u;
typedef __attribute__((ext_vector_type(2)))  unsigned v2u;

#define HIDN   168
#define OUTD   35
#define NT_H   11          // 176 / 16 N-tiles for hidden
#define KT_H   6           // 192 / 32 K-steps for hidden
#define NT_O   3           // 48 / 16 N-tiles for output
#define STRIDE 192         // staging row stride in halves (16B-multiple rows)
#define W2OFF  (11 * 512)                 // halves
#define W3OFF  (W2OFF + 66 * 512)
#define WTOT   (W3OFF + 18 * 512)         // 48640 halves = 95 KB

struct Params {
  const float* xyz;
  const float* pl[3][3];                  // [scale][xy,xz,yz], each (8,S,S)
  const float* w1[3]; const float* b1[3];
  const float* w2[3]; const float* b2[3];
  const float* w3[3]; const float* b3[3];
  float* out;
};

// A fragment (16x32 f16): lanes 0-15 -> M=0..15, K 0..7 & 16..23; lanes 16-31 -> K 8..15 & 24..31.
__device__ __forceinline__ v16h load_a_frag(const _Float16* stg, int kt, int lane) {
  const int m = lane & 15, g = lane >> 4;
  const _Float16* row = stg + m * STRIDE + kt * 32 + g * 8;
  union { v4u q[2]; v16h h; } u;
  u.q[0] = *(const v4u*)(row);            // K = kt*32 + g*8 .. +7
  u.q[1] = *(const v4u*)(row + 16);       // K = kt*32 + 16 + g*8 .. +7
  return u.h;
}

// B fragment pre-swizzled in LDS: 32 contiguous bytes per lane per tile.
__device__ __forceinline__ v16h load_b_frag(const _Float16* w, int tile, int lane) {
  return *(const v16h*)(w + tile * 512 + lane * 16);
}

__global__ __launch_bounds__(256) void triplane_mlp_kernel(Params P, int npts) {
  __shared__ __align__(16) _Float16 lds_w[WTOT];               // 97280 B
  __shared__ __align__(16) _Float16 lds_stg[8 * 16 * STRIDE];  // 49152 B

  const int tid  = threadIdx.x;
  const int wave = tid >> 5;
  const int lane = tid & 31;
  _Float16* stg  = lds_stg + wave * 16 * STRIDE;

  const int p0     = (blockIdx.x * 8 + wave) * 16;   // this wave's 16-point tile
  const bool active = p0 < npts;

  // Zero staging once (covers permanent pad cols 176..191 and initial 24..31).
  {
    v4u z = {};
    for (int i = lane; i < (16 * STRIDE) / 8; i += 32)
      *(v4u*)(stg + i * 8) = z;
  }

  // Lane roles: point pp = lane&15 (matches WMMA M), channel half g = lane>>4.
  const int pp = lane & 15;
  const int g  = lane >> 4;
  int pidx = p0 + pp; if (pidx >= npts) pidx = npts - 1;
  const float px = P.xyz[3 * pidx + 0];
  const float py = P.xyz[3 * pidx + 1];
  const float pz = P.xyz[3 * pidx + 2];
  const float nx = (px - (-2.f)) / 4.f * 2.f - 1.f;
  const float ny = (py - (-2.f)) / 4.f * 2.f - 1.f;
  const float nz = (pz - (-2.f)) / 4.f * 2.f - 1.f;

  v8f acc[NT_O];
  #pragma unroll
  for (int nt = 0; nt < NT_O; ++nt) acc[nt] = v8f{};

  const int SZ[3] = {128, 256, 512};

  for (int s = 0; s < 3; ++s) {
    __syncthreads();   // previous scale's weight reads done before refill

    // ---- cooperative fill: f32 weights -> f16 B-fragment-swizzled LDS ----
    {
      const float* w1g = P.w1[s];
      const float* w2g = P.w2[s];
      const float* w3g = P.w3[s];
      for (int idx = tid; idx < WTOT; idx += 256) {
        const int tile = idx >> 9;
        const int r    = idx & 511;
        const int ln   = r >> 4;
        const int e    = r & 15;
        int kt, nt, K, N; const float* src;
        if (tile < 11)      { kt = 0; nt = tile; K = 24; N = HIDN; src = w1g; }
        else if (tile < 77) { int t = tile - 11; kt = t / 11; nt = t - kt * 11; K = HIDN; N = HIDN; src = w2g; }
        else                { int t = tile - 77; kt = t / 3;  nt = t - kt * 3;  K = HIDN; N = OUTD; src = w3g; }
        const int k = kt * 32 + (e >> 3) * 16 + (ln >> 4) * 8 + (e & 7);      // B layout: table order (K 0..7 & 16..23 / 8..15 & 24..31 per lane group)
        const int n = nt * 16 + (ln & 15);
        const float v = (k < K && n < N) ? src[k * N + n] : 0.f;
        lds_w[idx] = (_Float16)v;
      }
    }
    __syncthreads();

    if (active) {
      const int S = SZ[s];

      // re-zero feat pad cols 24..31 (clobbered by h2 of previous scale)
      { v2u z = {}; *(v2u*)(stg + pp * STRIDE + 24 + g * 4) = z; }

      // ---- bilinear gather: 3 planes x 4 channels per lane -> staging cols 0..23 ----
      #pragma unroll
      for (int pl = 0; pl < 3; ++pl) {
        const float gx = (pl == 0) ? ny : nz;            // xy:ny  xz:nz  yz:nz
        const float gy = (pl == 2) ? ny : nx;            // xy:nx  xz:nx  yz:ny
        const float ix = (gx + 1.f) * 0.5f * (float)(S - 1);
        const float iy = (gy + 1.f) * 0.5f * (float)(S - 1);
        const float fx = floorf(ix), fy = floorf(iy);
        const float wx = ix - fx,    wy = iy - fy;
        const int x0 = (int)fx, y0 = (int)fy;
        const int x0c = min(max(x0, 0), S - 1);
        const int x1c = min(max(x0 + 1, 0), S - 1);
        const int y0c = min(max(y0, 0), S - 1);
        const int y1c = min(max(y0 + 1, 0), S - 1);
        const int SS  = S * S;
        const float* base = P.pl[s][pl] + (g * 4) * SS;
        const int i00 = y0c * S + x0c, i01 = y0c * S + x1c;
        const int i10 = y1c * S + x0c, i11 = y1c * S + x1c;
        #pragma unroll
        for (int j = 0; j < 4; ++j) {
          const float v00 = base[i00], v01 = base[i01];
          const float v10 = base[i10], v11 = base[i11];
          base += SS;
          const float top = v00 * (1.f - wx) + v01 * wx;
          const float bot = v10 * (1.f - wx) + v11 * wx;
          const float f   = top * (1.f - wy) + bot * wy;
          stg[pp * STRIDE + pl * 8 + g * 4 + j] = (_Float16)f;
        }
      }
      asm volatile("" ::: "memory");

      // ---- layer 1: feat(16x32) x w1(32x176), single K-step kept in regs ----
      const v16h afeat = load_a_frag(stg, 0, lane);
      asm volatile("" ::: "memory");
      const float* b1g = P.b1[s];
      const int ncol = lane & 15;
      #pragma unroll
      for (int nt = 0; nt < NT_H; ++nt) {
        v16h b = load_b_frag(lds_w, nt, lane);
        v8f d = {};
        d = __builtin_amdgcn_wmma_f32_16x16x32_f16(false, afeat, false, b,
                                                   (short)0, d, false, false);
        const int col = nt * 16 + ncol;
        const float bias = (col < HIDN) ? b1g[col] : 0.f;
        #pragma unroll
        for (int r = 0; r < 8; ++r) {
          float v = d[r] + bias;
          stg[(r + 8 * g) * STRIDE + col] = (_Float16)(v > 0.f ? v : 0.f);
        }
      }
      asm volatile("" ::: "memory");

      // ---- layer 2: h1(16x192) x w2(192x176); preload A so staging can be reused ----
      v16h a2[KT_H];
      #pragma unroll
      for (int kt = 0; kt < KT_H; ++kt) a2[kt] = load_a_frag(stg, kt, lane);
      asm volatile("" ::: "memory");
      const float* b2g = P.b2[s];
      #pragma unroll
      for (int nt = 0; nt < NT_H; ++nt) {
        v8f d = {};
        #pragma unroll
        for (int kt = 0; kt < KT_H; ++kt) {
          v16h b = load_b_frag(lds_w + W2OFF, kt * 11 + nt, lane);
          d = __builtin_amdgcn_wmma_f32_16x16x32_f16(false, a2[kt], false, b,
                                                     (short)0, d, false, false);
        }
        const int col = nt * 16 + ncol;
        const float bias = (col < HIDN) ? b2g[col] : 0.f;
        #pragma unroll
        for (int r = 0; r < 8; ++r) {
          float v = d[r] + bias;
          stg[(r + 8 * g) * STRIDE + col] = (_Float16)(v > 0.f ? v : 0.f);
        }
      }
      asm volatile("" ::: "memory");

      // ---- layer 3: h2(16x192) x w3(192x48), accumulate across scales ----
      #pragma unroll
      for (int kt = 0; kt < KT_H; ++kt) a2[kt] = load_a_frag(stg, kt, lane);
      asm volatile("" ::: "memory");
      const float* b3g = P.b3[s];
      #pragma unroll
      for (int nt = 0; nt < NT_O; ++nt) {
        v8f d = acc[nt];
        #pragma unroll
        for (int kt = 0; kt < KT_H; ++kt) {
          v16h b = load_b_frag(lds_w + W3OFF, kt * 3 + nt, lane);
          d = __builtin_amdgcn_wmma_f32_16x16x32_f16(false, a2[kt], false, b,
                                                     (short)0, d, false, false);
        }
        const int col = nt * 16 + ncol;
        const float bias = (col < OUTD) ? b3g[col] : 0.f;
        #pragma unroll
        for (int r = 0; r < 8; ++r) d[r] += bias;
        acc[nt] = d;
      }
    }
  }

  // ---- epilogue: raw / sigmoid(31..33) / x10(34), store f32 ----
  if (active) {
    #pragma unroll
    for (int nt = 0; nt < NT_O; ++nt) {
      const int col = nt * 16 + (lane & 15);
      if (col < OUTD) {
        #pragma unroll
        for (int r = 0; r < 8; ++r) {
          const int row = p0 + r + 8 * g;
          if (row < npts) {
            float v = acc[nt][r];
            if (col >= 31 && col < 34)      v = 1.f / (1.f + expf(-v));
            else if (col == 34)             v *= 10.f;
            P.out[(size_t)row * OUTD + col] = v;
          }
        }
      }
    }
  }
}

extern "C" void kernel_launch(void* const* d_in, const int* in_sizes, int n_in,
                              void* d_out, int out_size, void* d_ws, size_t ws_size,
                              hipStream_t stream) {
  (void)n_in; (void)out_size; (void)d_ws; (void)ws_size;
  Params p;
  p.xyz = (const float*)d_in[0];
  for (int i = 0; i < 3; ++i) {
    const int b = 1 + i * 9;
    p.pl[i][0] = (const float*)d_in[b + 0];  // xy
    p.pl[i][1] = (const float*)d_in[b + 1];  // xz
    p.pl[i][2] = (const float*)d_in[b + 2];  // yz
    p.w1[i] = (const float*)d_in[b + 3];
    p.b1[i] = (const float*)d_in[b + 4];
    p.w2[i] = (const float*)d_in[b + 5];
    p.b2[i] = (const float*)d_in[b + 6];
    p.w3[i] = (const float*)d_in[b + 7];
    p.b3[i] = (const float*)d_in[b + 8];
  }
  p.out = (float*)d_out;

  const int npts   = in_sizes[0] / 3;
  const int tiles  = (npts + 15) / 16;
  const int blocks = (tiles + 7) / 8;
  triplane_mlp_kernel<<<blocks, 256, 0, stream>>>(p, npts);
}
